// GRU_82446192214593
// MI455X (gfx1250) — matrix operand order repaired and checked by reading, not compiled, not run on hardware
//
#include <hip/hip_runtime.h>
#include <stdint.h>

#define BB   4
#define NN   8192
#define KK   32
#define HID  64
#define CH   128      // IN_CH = HID + INP
#define W1R  131      // IN_CH + 3
#define PT   16       // points per wave
#define WAVES 4       // waves per workgroup

typedef __attribute__((ext_vector_type(16))) __bf16       v16bf;
typedef __attribute__((ext_vector_type(8)))  float        v8f;
typedef __attribute__((ext_vector_type(8)))  unsigned int v8u;

__device__ __forceinline__ unsigned short f2bf(float f) {
  unsigned int u = __float_as_uint(f);
  u += 0x7fffu + ((u >> 16) & 1u);          // round-to-nearest-even
  return (unsigned short)(u >> 16);
}

__device__ __forceinline__ v16bf pack2(uint4 a, uint4 b) {
  v8u u;
  u[0]=a.x; u[1]=a.y; u[2]=a.z; u[3]=a.w;
  u[4]=b.x; u[5]=b.y; u[6]=b.z; u[7]=b.w;
  return __builtin_bit_cast(v16bf, u);
}

__device__ __forceinline__ v16bf ld_frag(const unsigned short* p) {
  const uint4* q = (const uint4*)p;
  return pack2(q[0], q[1]);
}

__device__ __forceinline__ float max8(v8f a) {
  float m0 = fmaxf(a[0], a[1]), m1 = fmaxf(a[2], a[3]);
  float m2 = fmaxf(a[4], a[5]), m3 = fmaxf(a[6], a[7]);
  return fmaxf(fmaxf(m0, m1), fmaxf(m2, m3));
}

// ---------------- prep kernels: swizzle weights into WMMA B-fragment order --

// B-fragment layout (bf16 32x16, one WMMA k-step x one 16-col tile):
//   lane L holds 16 halves: W[32*ks + (L<16?0:8) + t + (t<8?0:8)][16*ni + (L&15)], t=0..15
__global__ void prep_w1(const float* __restrict__ W1, unsigned short* __restrict__ w1f) {
  int t = blockIdx.x * blockDim.x + threadIdx.x;
  if (t >= 3 * 5 * 4 * 32) return;              // gate*5ks*4ni*32lane
  int lane = t & 31;
  int ni   = (t >> 5) & 3;
  int ks   = (t >> 7) % 5;
  int gate = (t >> 7) / 5;
  int rbase = ks * 32 + ((lane & 16) ? 8 : 0);
  int col   = (ni << 4) + (lane & 15);
#pragma unroll
  for (int tt = 0; tt < 16; ++tt) {
    int row = rbase + tt + (tt >= 8 ? 8 : 0);
    float v = (row < W1R) ? W1[((size_t)gate * W1R + row) * HID + col] : 0.f;
    w1f[(size_t)t * 16 + tt] = f2bf(v);
  }
}

__global__ void prep_w23(const float* __restrict__ W2, const float* __restrict__ W3,
                         unsigned short* __restrict__ w2f, unsigned short* __restrict__ w3f) {
  int t = blockIdx.x * blockDim.x + threadIdx.x;
  if (t >= 2 * 3 * 2 * 4 * 32) return;
  const float*   W = (t < 768) ? W2 : W3;
  unsigned short* o = (t < 768) ? w2f : w3f;
  int u = t & 767;
  int lane = u & 31;
  int ni   = (u >> 5) & 3;
  int ks   = (u >> 7) & 1;
  int gate = u >> 8;
  int rbase = ks * 32 + ((lane & 16) ? 8 : 0);
  int col   = (ni << 4) + (lane & 15);
#pragma unroll
  for (int tt = 0; tt < 16; ++tt) {
    int row = rbase + tt + (tt >= 8 ? 8 : 0);
    o[(size_t)u * 16 + tt] = f2bf(W[((size_t)gate * 64 + row) * 64 + col]);
  }
}

// ---------------- feature packing (fp32 -> bf16) ----------------------------

__global__ void pack_feat(const float* __restrict__ h, const float* __restrict__ x,
                          unsigned short* __restrict__ featA) {
  size_t i = (size_t)blockIdx.x * blockDim.x + threadIdx.x;   // B*N*128 threads
  size_t pid = i >> 7;
  int ch = (int)(i & 127);
  float v = (ch < 64) ? h[pid * 64 + ch] : x[pid * 64 + (ch - 64)];
  featA[i] = f2bf(v);
}

__global__ void pack_feat2(const unsigned short* __restrict__ rh,
                           unsigned short* __restrict__ featA) {
  size_t i = (size_t)blockIdx.x * blockDim.x + threadIdx.x;   // B*N*64 threads
  size_t pid = i >> 6;
  int ch = (int)(i & 63);
  featA[pid * 128 + ch] = rh[i];                               // upper half keeps x
}

// ---------------- main SetConv kernel ---------------------------------------
// PHASE 0: gates {z, r} sharing one gather; writes z (f32) and r*h (bf16).
// PHASE 1: gate q; reads z, writes final (1-z)*h + z*q.

template <int PHASE>
__global__ __launch_bounds__(WAVES * 32)
void setconv_kernel(const unsigned short* __restrict__ featA,
                    const unsigned short* __restrict__ w1f,
                    const unsigned short* __restrict__ w2f,
                    const unsigned short* __restrict__ w3f,
                    const float* __restrict__ b1,
                    const float* __restrict__ b2,
                    const float* __restrict__ b3,
                    const int*   __restrict__ neigh,
                    const float* __restrict__ edge,
                    const float* __restrict__ h_in,
                    float* __restrict__ zbuf,
                    unsigned short* __restrict__ rhbuf,
                    float* __restrict__ outp) {
  constexpr int NG = (PHASE == 0) ? 2 : 1;
  constexpr int G0 = (PHASE == 0) ? 0 : 2;

  __shared__ __align__(16) unsigned short w1s[NG * 5 * 4 * 32 * 16];
  __shared__ __align__(16) unsigned short ybuf[WAVES * NG * PT * HID];

  const int tid  = threadIdx.x;
  const int wave = tid >> 5;
  const int lane = tid & 31;
  const int lsub = lane & 15;
  const bool hi  = lane >= 16;

  // stage W1 fragments for our gates into LDS (reused for 64 points)
  {
    const unsigned int* s = (const unsigned int*)(w1f + (size_t)G0 * 5 * 4 * 32 * 16);
    unsigned int* d = (unsigned int*)w1s;
    const int cnt = NG * 5 * 4 * 32 * 16 / 2;
    for (int i = tid; i < cnt; i += WAVES * 32) d[i] = s[i];
  }
  __syncthreads();

  const v8f vzero = {0.f, 0.f, 0.f, 0.f, 0.f, 0.f, 0.f, 0.f};
  const int ptBase = (blockIdx.x * WAVES + wave) * PT;

  // ---------------- stage 1: gather + [32x160]x[160x64] WMMA + max-pool ------
  for (int p = 0; p < PT; ++p) {
    const int pid = ptBase + p;
    const int bb  = pid >> 13;                 // / N
    float ymax[NG][4];

#pragma unroll
    for (int mi = 0; mi < 2; ++mi) {
      const int row = mi * 16 + lsub;          // neighbor index 0..31
      const int idx = neigh[(size_t)pid * KK + row];
      const unsigned short* src = featA + ((size_t)bb * NN + idx) * CH;

      v8f acc[NG][4];
#pragma unroll
      for (int g = 0; g < NG; ++g)
#pragma unroll
        for (int ni = 0; ni < 4; ++ni) acc[g][ni] = vzero;

#pragma unroll
      for (int ks = 0; ks < 5; ++ks) {
        v16bf A;
        if (ks < 4) {                          // gathered channels: direct from L2
          const int c = ks * 32 + (hi ? 8 : 0);
          A = pack2(*(const uint4*)(src + c), *(const uint4*)(src + c + 16));
        } else {                               // edge features + zero pad (cols 128..159)
          uint4 e4 = {0, 0, 0, 0};
          uint4 z4 = {0, 0, 0, 0};
          if (!hi) {
            const float* e = edge + ((size_t)pid * KK + row) * 3;
            e4.x = (unsigned)f2bf(e[0]) | ((unsigned)f2bf(e[1]) << 16);
            e4.y = (unsigned)f2bf(e[2]);
          }
          A = pack2(e4, z4);
        }
#pragma unroll
        for (int g = 0; g < NG; ++g) {
#pragma unroll
          for (int ni = 0; ni < 4; ++ni) {
            v16bf Bm = ld_frag(&w1s[(((g * 5 + ks) * 4 + ni) * 32 + lane) * 16]);
            acc[g][ni] = __builtin_amdgcn_wmma_f32_16x16x32_bf16(
                false, A, false, Bm, (short)0, acc[g][ni], false, false);
          }
        }
      }
      // max over the 16 rows of this M-tile, then across mi tiles
#pragma unroll
      for (int g = 0; g < NG; ++g)
#pragma unroll
        for (int ni = 0; ni < 4; ++ni) {
          float m = max8(acc[g][ni]);
          m = fmaxf(m, __int_as_float(
                  __builtin_amdgcn_ds_swizzle(__float_as_int(m), 0x401f)));  // SWAPX16
          ymax[g][ni] = mi ? fmaxf(ymax[g][ni], m) : m;
        }
    }

    // bias + leaky (commutes with max), store y row for this point
#pragma unroll
    for (int g = 0; g < NG; ++g)
#pragma unroll
      for (int ni = 0; ni < 4; ++ni) {
        const int ch = (ni << 4) + lsub;
        float yv = ymax[g][ni] + b1[(G0 + g) * HID + ch];
        yv = (yv > 0.f) ? yv : 0.1f * yv;
        if (!hi) ybuf[((wave * NG + g) * PT + p) * HID + ch] = f2bf(yv);
      }
  }

  // ---------------- stages 2+3: [16pts x 64] x [64x64] twice -----------------
#pragma unroll
  for (int gl = 0; gl < NG; ++gl) {
    const int gate = G0 + gl;
    unsigned short* yb = &ybuf[(size_t)(wave * NG + gl) * PT * HID];

    v16bf A2[2];
#pragma unroll
    for (int ks = 0; ks < 2; ++ks) {
      const int c = ks * 32 + (hi ? 8 : 0);
      const unsigned short* r = yb + lsub * HID + c;   // M = point = lsub
      A2[ks] = pack2(*(const uint4*)r, *(const uint4*)(r + 16));
    }
    v8f acc2[4] = {vzero, vzero, vzero, vzero};
#pragma unroll
    for (int ks = 0; ks < 2; ++ks)
#pragma unroll
      for (int ni = 0; ni < 4; ++ni) {
        v16bf Bm = ld_frag(w2f + (size_t)(((gate * 2 + ks) * 4 + ni) * 32 + lane) * 16);
        acc2[ni] = __builtin_amdgcn_wmma_f32_16x16x32_bf16(
            false, A2[ks], false, Bm, (short)0, acc2[ni], false, false);
      }
    // t = leaky(acc2 + b2) -> overwrite wave-private buffer
#pragma unroll
    for (int ni = 0; ni < 4; ++ni) {
      const int ch = (ni << 4) + lsub;
      const float bb2 = b2[gate * HID + ch];
#pragma unroll
      for (int v = 0; v < 8; ++v) {
        const int prow = hi ? v + 8 : v;               // D layout: M = v (+8 for hi half)
        float val = acc2[ni][v] + bb2;
        val = (val > 0.f) ? val : 0.1f * val;
        yb[prow * HID + ch] = f2bf(val);
      }
    }

    v16bf A3[2];
#pragma unroll
    for (int ks = 0; ks < 2; ++ks) {
      const int c = ks * 32 + (hi ? 8 : 0);
      const unsigned short* r = yb + lsub * HID + c;
      A3[ks] = pack2(*(const uint4*)r, *(const uint4*)(r + 16));
    }
    v8f acc3[4] = {vzero, vzero, vzero, vzero};
#pragma unroll
    for (int ks = 0; ks < 2; ++ks)
#pragma unroll
      for (int ni = 0; ni < 4; ++ni) {
        v16bf Bm = ld_frag(w3f + (size_t)(((gate * 2 + ks) * 4 + ni) * 32 + lane) * 16);
        acc3[ni] = __builtin_amdgcn_wmma_f32_16x16x32_bf16(
            false, A3[ks], false, Bm, (short)0, acc3[ni], false, false);
      }

    // activations + outputs
#pragma unroll
    for (int ni = 0; ni < 4; ++ni) {
      const int ch = (ni << 4) + lsub;
      const float bb3 = b3[gate * HID + ch];
#pragma unroll
      for (int v = 0; v < 8; ++v) {
        const int prow = hi ? v + 8 : v;
        const int pid  = ptBase + prow;
        const size_t gi = (size_t)pid * HID + ch;
        float val = acc3[ni][v] + bb3;
        if (PHASE == 0) {
          float s = 1.f / (1.f + __expf(-val));
          if (gl == 0) zbuf[gi] = s;                      // z
          else         rhbuf[gi] = f2bf(s * h_in[gi]);    // r*h
        } else {
          float q = tanhf(val);
          float z = zbuf[gi];
          outp[gi] = (1.f - z) * h_in[gi] + z * q;
        }
      }
    }
  }
}

// ---------------- launcher ---------------------------------------------------

extern "C" void kernel_launch(void* const* d_in, const int* in_sizes, int n_in,
                              void* d_out, int out_size, void* d_ws, size_t ws_size,
                              hipStream_t stream) {
  (void)in_sizes; (void)n_in; (void)out_size; (void)ws_size;
  const float* h    = (const float*)d_in[0];
  const float* x    = (const float*)d_in[1];
  // d_in[2] = c, unused by the reference GRU body
  const float* W1   = (const float*)d_in[3];
  const float* b1   = (const float*)d_in[4];
  const float* W2   = (const float*)d_in[5];
  const float* b2   = (const float*)d_in[6];
  const float* W3   = (const float*)d_in[7];
  const float* b3   = (const float*)d_in[8];
  const int*   ngh  = (const int*)d_in[9];
  const float* edge = (const float*)d_in[10];
  float* outp = (float*)d_out;

  char* ws = (char*)d_ws;
  size_t off = 0;
  unsigned short* featA = (unsigned short*)(ws + off); off += (size_t)BB * NN * CH * 2;
  float*          zbuf  = (float*)(ws + off);          off += (size_t)BB * NN * HID * 4;
  unsigned short* rh    = (unsigned short*)(ws + off); off += (size_t)BB * NN * HID * 2;
  unsigned short* w1f   = (unsigned short*)(ws + off); off += (size_t)3 * 5 * 4 * 32 * 16 * 2;
  unsigned short* w2f   = (unsigned short*)(ws + off); off += (size_t)3 * 2 * 4 * 32 * 16 * 2;
  unsigned short* w3f   = (unsigned short*)(ws + off); off += (size_t)3 * 2 * 4 * 32 * 16 * 2;

  prep_w1 <<<dim3(15), dim3(128), 0, stream>>>(W1, w1f);
  prep_w23<<<dim3(12), dim3(128), 0, stream>>>(W2, W3, w2f, w3f);
  pack_feat<<<dim3(BB * NN * CH / 256), dim3(256), 0, stream>>>(h, x, featA);

  const int nblk = BB * NN / (WAVES * PT);   // 512
  setconv_kernel<0><<<dim3(nblk), dim3(WAVES * 32), 0, stream>>>(
      featA, w1f, w2f, w3f, b1, b2, b3, ngh, edge, h, zbuf, rh, outp);

  pack_feat2<<<dim3(BB * NN * HID / 256), dim3(256), 0, stream>>>(rh, featA);

  setconv_kernel<1><<<dim3(nblk), dim3(WAVES * 32), 0, stream>>>(
      featA, w1f, w2f, w3f, b1, b2, b3, ngh, edge, h, zbuf, rh, outp);
}
